// CoreMLOptimizedCosmicFish_51625506898302
// MI455X (gfx1250) — compile-verified, matrix operand order checked by reading, not run on hardware
//
#include <hip/hip_runtime.h>

// ---- model dims ----
#define Bsz  2
#define Tseq 512
#define Cdim 1024
#define Hn   16
#define KVn  4
#define HDn  64
#define Lnum 8
#define Vnum 32000
#define FFn  4096
#define QKVn (Cdim + 2*KVn*HDn)   // 1536
#define Mtok (Bsz*Tseq)           // 1024
#define REP  (Hn/KVn)             // 4

typedef __attribute__((ext_vector_type(16))) __bf16 v16bf;
typedef __attribute__((ext_vector_type(8)))  float  v8f;
typedef __attribute__((ext_vector_type(2)))  __bf16 v2bf;
typedef __attribute__((ext_vector_type(4)))  __bf16 v4bf;

union Frag16 { v16bf v; unsigned short u[16]; uint4 q[2]; };

// hardware f32 -> bf16 (lowers to native v_cvt on gfx1250)
__device__ __forceinline__ unsigned short f2bf(float f) {
  return __builtin_bit_cast(unsigned short, (__bf16)f);
}
// packed pair -> one v_cvt_pk_bf16_f32
__device__ __forceinline__ unsigned pack2bf(float a, float b) {
#if __has_builtin(__builtin_amdgcn_cvt_pk_bf16_f32)
  return __builtin_bit_cast(unsigned, __builtin_amdgcn_cvt_pk_bf16_f32(a, b));
#else
  v2bf v; v[0] = (__bf16)a; v[1] = (__bf16)b;
  return __builtin_bit_cast(unsigned, v);
#endif
}
// packed quad -> two v_cvt_pk_bf16_f32, stored as uint2 (for ds_store_b64)
__device__ __forceinline__ uint2 pack4bf(float a, float b, float c, float d) {
  uint2 r; r.x = pack2bf(a, b); r.y = pack2bf(c, d); return r;
}
__device__ __forceinline__ float geluf(float x) {
  return 0.5f * x * (1.0f + tanhf(0.7978845608028654f * (x + 0.044715f * x * x * x)));
}

// ---------------- embedding gather ----------------
__global__ void embed_k(const int* __restrict__ ids, const float* __restrict__ wte,
                        float* __restrict__ x) {
  int tok = blockIdx.x;
  int id  = ids[tok];
  for (int c = threadIdx.x; c < Cdim; c += blockDim.x)
    x[(size_t)tok*Cdim + c] = wte[(size_t)id*Cdim + c];
}

// ---------------- LayerNorm ----------------
__global__ __launch_bounds__(256) void ln_k(const float* __restrict__ x,
                                            const float* __restrict__ w,
                                            const float* __restrict__ b,
                                            float* __restrict__ out) {
  __shared__ float s1[256], s2[256];
  int row = blockIdx.x, tid = threadIdx.x;
  const float* xr = x + (size_t)row*Cdim;
  float sum = 0.f, sq = 0.f;
  for (int c = tid; c < Cdim; c += 256) { float v = xr[c]; sum += v; sq += v*v; }
  s1[tid] = sum; s2[tid] = sq; __syncthreads();
  for (int st = 128; st > 0; st >>= 1) {
    if (tid < st) { s1[tid] += s1[tid+st]; s2[tid] += s2[tid+st]; }
    __syncthreads();
  }
  float mean = s1[0] * (1.0f/Cdim);
  float var  = s2[0] * (1.0f/Cdim) - mean*mean;
  float rs   = rsqrtf(var + 1e-5f);
  float* orow = out + (size_t)row*Cdim;
  for (int c = tid; c < Cdim; c += 256)
    orow[c] = (xr[c]-mean)*rs*w[c] + b[c];
}

// ---------------- generic bf16-WMMA GEMM, double-buffered LDS ----------------
// D[M,N] = epi(A[M,K] @ B[K,N] + bias [+ R])
// epi: 0 none, 1 +bias, 2 +bias,gelu, 3 +bias,+residual R
#define BM 128
#define BN 128
#define BK 32
#define LDSS 40    // padded row stride (ushort) to spread LDS banks

__global__ __launch_bounds__(256) void gemm_bf16_k(
    const float* __restrict__ A, const float* __restrict__ Bm,
    const float* __restrict__ bias, const float* __restrict__ R,
    float* __restrict__ D, int M, int N, int K, int epi) {
  __shared__ __align__(16) unsigned short As[2][BM][LDSS];
  __shared__ __align__(16) unsigned short Bs[2][BN][LDSS];   // transposed: Bs[b][n][k]
  int tid  = threadIdx.x;
  int wave = tid >> 5, lane = tid & 31;
  int wm = wave >> 2, wn = wave & 3;          // 2x4 wave grid -> 128x128 tile
  int r = lane & 15, half = lane >> 4;
  int blockM = blockIdx.y * BM, blockN = blockIdx.x * BN;

  v8f acc[4][2];
  #pragma unroll
  for (int i = 0; i < 4; i++)
    #pragma unroll
    for (int j = 0; j < 2; j++)
      #pragma unroll
      for (int e = 0; e < 8; e++) acc[i][j][e] = 0.f;

  float4 ra[4];
  float  rbv[4][4];
  int nB  = tid & 127;        // column within B tile (coalesced across lanes)
  int kq0 = tid >> 7;         // 0..1 (quad-of-k group)

  auto fetch = [&](int k0) {
    #pragma unroll
    for (int it = 0; it < 4; it++) {
      int idx = tid*4 + it*1024;
      ra[it] = *reinterpret_cast<const float4*>(
          &A[(size_t)(blockM + (idx >> 5))*K + k0 + (idx & 31)]);
    }
    #pragma unroll
    for (int g = 0; g < 4; g++) {
      int kq = kq0 + g*2;     // 0..7 -> k = kq*4 + j
      #pragma unroll
      for (int j = 0; j < 4; j++)
        rbv[g][j] = Bm[(size_t)(k0 + kq*4 + j)*N + blockN + nB];
    }
  };
  auto stage = [&](int buf) {
    #pragma unroll
    for (int it = 0; it < 4; it++) {
      int idx = tid*4 + it*1024;
      int rowa = idx >> 5, cola = idx & 31;
      *reinterpret_cast<uint2*>(&As[buf][rowa][cola]) =
          pack4bf(ra[it].x, ra[it].y, ra[it].z, ra[it].w);
    }
    #pragma unroll
    for (int g = 0; g < 4; g++) {
      int kq = kq0 + g*2;
      *reinterpret_cast<uint2*>(&Bs[buf][nB][kq*4]) =
          pack4bf(rbv[g][0], rbv[g][1], rbv[g][2], rbv[g][3]);
    }
  };

  int nk = K / BK;
  fetch(0);
  stage(0);
  for (int kt = 0; kt < nk; kt++) {
    __syncthreads();                       // LDS[kt&1] ready for all waves
    if (kt + 1 < nk) fetch((kt + 1) * BK); // issue next tile's global loads early
    int cur = kt & 1;

    Frag16 bf[2];
    #pragma unroll
    for (int ni = 0; ni < 2; ni++) {
      int colb = wn*32 + ni*16 + r;
      bf[ni].q[0] = *reinterpret_cast<const uint4*>(&Bs[cur][colb][half*8]);
      bf[ni].q[1] = *reinterpret_cast<const uint4*>(&Bs[cur][colb][16 + half*8]);
    }
    #pragma unroll
    for (int mi = 0; mi < 4; mi++) {
      Frag16 af;
      int rowa = wm*64 + mi*16 + r;
      af.q[0] = *reinterpret_cast<const uint4*>(&As[cur][rowa][half*8]);
      af.q[1] = *reinterpret_cast<const uint4*>(&As[cur][rowa][16 + half*8]);
      #pragma unroll
      for (int ni = 0; ni < 2; ni++)
        acc[mi][ni] = __builtin_amdgcn_wmma_f32_16x16x32_bf16(
            false, af.v, false, bf[ni].v, (short)0, acc[mi][ni], false, false);
    }
    // stores go to the other buffer; loop-top barrier guarantees nobody still reads it
    if (kt + 1 < nk) stage((kt + 1) & 1);
  }

  // epilogue: VGPR i holds M = half*8+i, N = r within each 16x16 tile
  #pragma unroll
  for (int mi = 0; mi < 4; mi++)
    #pragma unroll
    for (int ni = 0; ni < 2; ni++)
      #pragma unroll
      for (int i = 0; i < 8; i++) {
        int row = blockM + wm*64 + mi*16 + half*8 + i;
        int col = blockN + wn*32 + ni*16 + r;
        if (row < M && col < N) {
          float v = acc[mi][ni][i];
          if (epi >= 1 && bias) v += bias[col];
          if (epi == 2) v = geluf(v);
          if (epi == 3) v += R[(size_t)row*N + col];
          D[(size_t)row*N + col] = v;
        }
      }
}

// ---------------- RoPE + GQA split: qkv fp32 -> Q/K bf16, V transposed bf16 ----------------
// Vt layout: [B*KVn][HDn][Tseq]  (so PV's B-fragment is a contiguous 2x b128 load)
__global__ void rope_split_k(const float* __restrict__ qkv,
                             const float* __restrict__ fcos, const float* __restrict__ fsin,
                             unsigned short* __restrict__ Qb,
                             unsigned short* __restrict__ Kb,
                             unsigned short* __restrict__ Vt) {
  int tok = blockIdx.x;            // 0..Mtok-1
  int slot = blockIdx.y;           // 0..H+2KV-1
  int d = threadIdx.x;             // 0..HD-1
  int b = tok / Tseq, t = tok % Tseq;
  const float* rowp = qkv + (size_t)tok * QKVn;
  if (slot < Hn) {
    const float* src = rowp + slot*HDn;
    int p = d >> 1;
    float c = fcos[t*(HDn/2) + p], s = fsin[t*(HDn/2) + p];
    float e = src[2*p], o = src[2*p+1];
    float v = (d & 1) ? (e*s + o*c) : (e*c - o*s);
    Qb[((size_t)(b*Hn + slot)*Tseq + t)*HDn + d] = f2bf(v);
  } else if (slot < Hn + KVn) {
    int kh = slot - Hn;
    const float* src = rowp + Cdim + kh*HDn;
    int p = d >> 1;
    float c = fcos[t*(HDn/2) + p], s = fsin[t*(HDn/2) + p];
    float e = src[2*p], o = src[2*p+1];
    float v = (d & 1) ? (e*s + o*c) : (e*c - o*s);
    Kb[((size_t)(b*KVn + kh)*Tseq + t)*HDn + d] = f2bf(v);
  } else {
    int kh = slot - Hn - KVn;
    const float* src = rowp + Cdim + KVn*HDn + kh*HDn;
    Vt[((size_t)(b*KVn + kh)*HDn + d)*Tseq + t] = f2bf(src[d]);
  }
}

// ---------------- attention scores: S = Q K^T * scale (causal tiles only) ----------------
__global__ __launch_bounds__(32) void attn_scores_k(const unsigned short* __restrict__ Qb,
                                                    const unsigned short* __restrict__ Kb,
                                                    float* __restrict__ scores) {
  int xt = blockIdx.x, yt = blockIdx.y, bh = blockIdx.z;
  if (xt > yt) return;                         // fully masked tile
  int b = bh / Hn, h = bh % Hn, kh = h / REP;
  int lane = threadIdx.x, r = lane & 15, half = lane >> 4;
  const unsigned short* qrow = Qb + ((size_t)(b*Hn  + h )*Tseq + yt*16 + r) * HDn;
  const unsigned short* krow = Kb + ((size_t)(b*KVn + kh)*Tseq + xt*16 + r) * HDn;
  v8f acc;
  #pragma unroll
  for (int i = 0; i < 8; i++) acc[i] = 0.f;
  #pragma unroll
  for (int ks = 0; ks < HDn; ks += 32) {
    Frag16 a, bfr;
    a.q[0]   = *reinterpret_cast<const uint4*>(qrow + ks + half*8);
    a.q[1]   = *reinterpret_cast<const uint4*>(qrow + ks + 16 + half*8);
    bfr.q[0] = *reinterpret_cast<const uint4*>(krow + ks + half*8);
    bfr.q[1] = *reinterpret_cast<const uint4*>(krow + ks + 16 + half*8);
    acc = __builtin_amdgcn_wmma_f32_16x16x32_bf16(false, a.v, false, bfr.v, (short)0, acc, false, false);
  }
  const float scale = 0.125f;   // 1/sqrt(64)
  #pragma unroll
  for (int i = 0; i < 8; i++) {
    int q = yt*16 + half*8 + i;
    int c = xt*16 + r;
    scores[((size_t)bh*Tseq + q)*Tseq + c] = acc[i] * scale;
  }
}

// ---------------- causal softmax (one wave per row) ----------------
__global__ __launch_bounds__(256) void softmax_k(const float* __restrict__ scores,
                                                 unsigned short* __restrict__ P) {
  int wave = threadIdx.x >> 5, lane = threadIdx.x & 31;
  int rowg = blockIdx.x * 8 + wave;            // 0..B*H*T-1
  int bh = rowg >> 9, rr = rowg & (Tseq-1);
  const float* srow = scores + ((size_t)bh*Tseq + rr)*Tseq;
  unsigned short* prow = P + ((size_t)bh*Tseq + rr)*Tseq;
  float v[16]; float m = -INFINITY;
  #pragma unroll
  for (int j = 0; j < 16; j++) {
    int c = lane + j*32;
    v[j] = (c <= rr) ? srow[c] : -INFINITY;
    m = fmaxf(m, v[j]);
  }
  #pragma unroll
  for (int off = 16; off >= 1; off >>= 1) m = fmaxf(m, __shfl_xor(m, off, 32));
  float sum = 0.f;
  #pragma unroll
  for (int j = 0; j < 16; j++) {
    int c = lane + j*32;
    v[j] = (c <= rr) ? __expf(v[j] - m) : 0.f;
    sum += v[j];
  }
  #pragma unroll
  for (int off = 16; off >= 1; off >>= 1) sum += __shfl_xor(sum, off, 32);
  float inv = 1.0f / sum;
  #pragma unroll
  for (int j = 0; j < 16; j++) prow[lane + j*32] = f2bf(v[j] * inv);
}

// ---------------- O = P @ V (V pre-transposed), written directly as [B,T,C] ----------------
__global__ __launch_bounds__(32) void attn_pv_k(const unsigned short* __restrict__ P,
                                                const unsigned short* __restrict__ Vt,
                                                float* __restrict__ obuf) {
  int xt = blockIdx.x;   // HD/16 tiles
  int yt = blockIdx.y;   // T/16 tiles
  int bh = blockIdx.z;
  int b = bh / Hn, h = bh % Hn, kh = h / REP;
  int lane = threadIdx.x, r = lane & 15, half = lane >> 4;
  const unsigned short* prow = P + ((size_t)bh*Tseq + yt*16 + r)*Tseq;
  int n = xt*16 + r;
  const unsigned short* vtrow = Vt + ((size_t)(b*KVn + kh)*HDn + n)*Tseq; // row n of V^T
  v8f acc;
  #pragma unroll
  for (int i = 0; i < 8; i++) acc[i] = 0.f;
  int kmax = ((yt*16 + 16) + 31) & ~31;        // causal: P is zero beyond row
  for (int k0 = 0; k0 < kmax; k0 += 32) {
    Frag16 a, bfr;
    a.q[0]   = *reinterpret_cast<const uint4*>(prow  + k0 + half*8);
    a.q[1]   = *reinterpret_cast<const uint4*>(prow  + k0 + 16 + half*8);
    bfr.q[0] = *reinterpret_cast<const uint4*>(vtrow + k0 + half*8);
    bfr.q[1] = *reinterpret_cast<const uint4*>(vtrow + k0 + 16 + half*8);
    acc = __builtin_amdgcn_wmma_f32_16x16x32_bf16(false, a.v, false, bfr.v, (short)0, acc, false, false);
  }
  #pragma unroll
  for (int i = 0; i < 8; i++) {
    int t = yt*16 + half*8 + i;
    obuf[((size_t)(b*Tseq + t))*Cdim + h*HDn + xt*16 + r] = acc[i];
  }
}

// ---------------- host orchestration ----------------
extern "C" void kernel_launch(void* const* d_in, const int* in_sizes, int n_in,
                              void* d_out, int out_size, void* d_ws, size_t ws_size,
                              hipStream_t stream) {
  (void)in_sizes; (void)n_in; (void)out_size; (void)ws_size;
  const int*   ids  = (const int*)  d_in[0];
  const float* wte  = (const float*)d_in[1];
  const float* fcos = (const float*)d_in[2];
  const float* fsin = (const float*)d_in[3];
  const float* ln1w = (const float*)d_in[4];
  const float* ln1b = (const float*)d_in[5];
  const float* wqkv = (const float*)d_in[6];
  const float* bqkv = (const float*)d_in[7];
  const float* wproj= (const float*)d_in[8];
  const float* bproj= (const float*)d_in[9];
  const float* ln2w = (const float*)d_in[10];
  const float* ln2b = (const float*)d_in[11];
  const float* wfc  = (const float*)d_in[12];
  const float* bfc  = (const float*)d_in[13];
  const float* wout = (const float*)d_in[14];
  const float* bout = (const float*)d_in[15];
  const float* lnfw = (const float*)d_in[16];
  const float* lnfb = (const float*)d_in[17];
  const float* wlm  = (const float*)d_in[18];

  char* p = (char*)d_ws;
  auto alloc = [&](size_t bytes) -> char* {
    char* q = p; p += (bytes + 255) & ~(size_t)255; return q;
  };
  float* x    = (float*)alloc((size_t)Mtok*Cdim*4);
  float* hbuf = (float*)alloc((size_t)Mtok*Cdim*4);
  float* qkv  = (float*)alloc((size_t)Mtok*QKVn*4);
  float* ff   = (float*)alloc((size_t)Mtok*FFn*4);
  float* obuf = (float*)alloc((size_t)Mtok*Cdim*4);
  unsigned short* Qb = (unsigned short*)alloc((size_t)Bsz*Hn*Tseq*HDn*2);
  unsigned short* Kb = (unsigned short*)alloc((size_t)Bsz*KVn*Tseq*HDn*2);
  unsigned short* Vt = (unsigned short*)alloc((size_t)Bsz*KVn*Tseq*HDn*2);
  float* scores = (float*)alloc((size_t)Bsz*Hn*Tseq*Tseq*4);
  unsigned short* P = (unsigned short*)alloc((size_t)Bsz*Hn*Tseq*Tseq*2);

  embed_k<<<Mtok, 256, 0, stream>>>(ids, wte, x);

  for (int l = 0; l < Lnum; l++) {
    ln_k<<<Mtok, 256, 0, stream>>>(x, ln1w + (size_t)l*Cdim, ln1b + (size_t)l*Cdim, hbuf);
    gemm_bf16_k<<<dim3(QKVn/128, Mtok/128), 256, 0, stream>>>(
        hbuf, wqkv + (size_t)l*Cdim*QKVn, bqkv + (size_t)l*QKVn, nullptr, qkv,
        Mtok, QKVn, Cdim, 1);
    rope_split_k<<<dim3(Mtok, Hn + 2*KVn), HDn, 0, stream>>>(qkv, fcos, fsin, Qb, Kb, Vt);
    attn_scores_k<<<dim3(Tseq/16, Tseq/16, Bsz*Hn), 32, 0, stream>>>(Qb, Kb, scores);
    softmax_k<<<(Bsz*Hn*Tseq)/8, 256, 0, stream>>>(scores, P);
    attn_pv_k<<<dim3(HDn/16, Tseq/16, Bsz*Hn), 32, 0, stream>>>(P, Vt, obuf);
    gemm_bf16_k<<<dim3(Cdim/128, Mtok/128), 256, 0, stream>>>(
        obuf, wproj + (size_t)l*Cdim*Cdim, bproj + (size_t)l*Cdim, x, x,
        Mtok, Cdim, Cdim, 3);
    ln_k<<<Mtok, 256, 0, stream>>>(x, ln2w + (size_t)l*Cdim, ln2b + (size_t)l*Cdim, hbuf);
    gemm_bf16_k<<<dim3(FFn/128, Mtok/128), 256, 0, stream>>>(
        hbuf, wfc + (size_t)l*Cdim*FFn, bfc + (size_t)l*FFn, nullptr, ff,
        Mtok, FFn, Cdim, 2);
    gemm_bf16_k<<<dim3(Cdim/128, Mtok/128), 256, 0, stream>>>(
        ff, wout + (size_t)l*FFn*Cdim, bout + (size_t)l*Cdim, x, x,
        Mtok, Cdim, FFn, 3);
  }

  ln_k<<<Mtok, 256, 0, stream>>>(x, lnfw, lnfb, hbuf);
  gemm_bf16_k<<<dim3(Vnum/128, Mtok/128), 256, 0, stream>>>(
      hbuf, wlm, nullptr, nullptr, (float*)d_out, Mtok, Vnum, Cdim, 0);
}
